// SelfAttention_21406117004050
// MI455X (gfx1250) — matrix operand order repaired and checked by reading, not compiled, not run on hardware
//
#include <hip/hip_runtime.h>
#include <math.h>

typedef __attribute__((ext_vector_type(16))) _Float16 v16h;
typedef __attribute__((ext_vector_type(8)))  float    v8f;

#define N_TOK   4096      // 64*64 tokens
#define C_DIM   128       // channels (attention inner dim)
#define J_TILE  128       // j-columns per workgroup (8 waves x 16)
#define I_CHUNK 64        // i (key) rows per flash step (4 S-tiles, 2 PV K-steps)
#define NCHUNK  (N_TOK / I_CHUNK)
#define RES_SCALE 0.002f

// LDS tiles, double buffered, filled by GLOBAL_LOAD_ASYNC_TO_LDS_B64.
// Pitches 8B-aligned for async B64 destinations:
//  tileCI[c][i]: pitch 68 halfs = 136B (34-bank stride, 2-way conflict)
//  tileIC[i][c]: pitch 132 halfs = 264B (66-bank stride, 2-way conflict)
#define CI_PITCH 68
#define IC_PITCH 132
#define CI_BYTES (C_DIM * CI_PITCH * 2)    // 17408
#define IC_BYTES (I_CHUNK * IC_PITCH * 2)  // 16896
#define BUF_BYTES (CI_BYTES + IC_BYTES)    // 34304
#define BUF_HALF  (BUF_BYTES / 2)

// ---------------------------------------------------------------------------
// Kernel 1: one-shot fp32 -> f16 conversion of F into both orientations.
//   fh [b][c][n]  (c-major: Q frags + PV-matmul tile)
//   fht[b][n][c]  (i-major: K^T-matmul tile)
// ---------------------------------------------------------------------------
__global__ __launch_bounds__(256, 1)
void convert_kernel(const float* __restrict__ x,
                    _Float16* __restrict__ fh, _Float16* __restrict__ fht) {
  __shared__ _Float16 t[C_DIM * 65];
  const int b  = blockIdx.y;
  const int n0 = blockIdx.x * 64;
  const float* xb = x + (size_t)b * C_DIM * N_TOK;
  _Float16* fhb  = fh  + (size_t)b * C_DIM * N_TOK;
  _Float16* fhtb = fht + (size_t)b * N_TOK * C_DIM;
  #pragma unroll
  for (int r = 0; r < 32; ++r) {                // 128c x 64n tile, coalesced reads
    int e = threadIdx.x + r * 256;
    int c = e >> 6, n = e & 63;
    _Float16 h = (_Float16)xb[(size_t)c * N_TOK + n0 + n];
    fhb[(size_t)c * N_TOK + n0 + n] = h;
    t[c * 65 + n] = h;
  }
  __syncthreads();
  #pragma unroll
  for (int r = 0; r < 32; ++r) {                // coalesced transposed writes
    int e = threadIdx.x + r * 256;
    int n = e >> 7, c = e & 127;
    fhtb[(size_t)(n0 + n) * C_DIM + c] = t[c * 65 + n];
  }
}

// ---------------------------------------------------------------------------
// Async staging: 16 GLOBAL_LOAD_ASYNC_TO_LDS_B64 per wave per chunk (32KB/WG).
// ---------------------------------------------------------------------------
__device__ __forceinline__
void stage_async(unsigned ldsbase, const _Float16* fhb, const _Float16* fhtb,
                 int wave, int lane, int i0) {
  // CI tile: wave w covers c rows [w*16, w*16+16); each instr: 2 rows x 128B
  #pragma unroll
  for (int t = 0; t < 8; ++t) {
    int c = wave * 16 + t * 2 + (lane >> 4);
    unsigned lds  = ldsbase + (unsigned)(c * (CI_PITCH * 2) + (lane & 15) * 8);
    unsigned voff = (unsigned)(c * (N_TOK * 2) + i0 * 2 + (lane & 15) * 8);
    asm volatile("global_load_async_to_lds_b64 %0, %1, %2"
                 :: "v"(lds), "v"(voff), "s"(fhb) : "memory");
  }
  // IC tile: wave w covers i rows [w*8, w*8+8); each instr: 1 row x 256B
  #pragma unroll
  for (int t = 0; t < 8; ++t) {
    int i = wave * 8 + t;
    unsigned lds  = ldsbase + (unsigned)(CI_BYTES + i * (IC_PITCH * 2) + lane * 8);
    unsigned voff = (unsigned)((i0 + i) * (C_DIM * 2) + lane * 8);
    asm volatile("global_load_async_to_lds_b64 %0, %1, %2"
                 :: "v"(lds), "v"(voff), "s"(fhtb) : "memory");
  }
}

// ---------------------------------------------------------------------------
// Kernel 2: fused flash self-attention, f16 WMMA, f32 accumulate.
// ---------------------------------------------------------------------------
__global__ __launch_bounds__(256, 1)
void fa_selfattn_kernel(const float* __restrict__ x,
                        const _Float16* __restrict__ fh,
                        const _Float16* __restrict__ fht,
                        float* __restrict__ out) {
  __shared__ __align__(16) _Float16 smem[2 * BUF_HALF];

  const int tid  = threadIdx.x;
  const int wave = tid >> 5;
  const int lane = tid & 31;
  const int jn   = lane & 15;   // N index inside 16x16 tile (same for lane, lane^16)
  const int hi   = lane >> 4;   // half-wave id
  const int b    = blockIdx.y;
  const int j    = blockIdx.x * J_TILE + wave * 16 + jn;

  const float*    xb   = x   + (size_t)b * C_DIM * N_TOK;
  const _Float16* fhb  = fh  + (size_t)b * C_DIM * N_TOK;
  const _Float16* fhtb = fht + (size_t)b * N_TOK * C_DIM;
  const unsigned  smembase = (unsigned)(uintptr_t)&smem[0];

  // ---- Q fragments: B-matrix layout; lanes 0-15 hold K=0..15, 16-31 K=16..31
  v16h qf[4];
  #pragma unroll
  for (int kk = 0; kk < 4; ++kk)
    #pragma unroll
    for (int k = 0; k < 16; ++k)
      qf[kk][k] = fhb[(size_t)(kk * 32 + hi * 16 + k) * N_TOK + j];

  // ---- online-softmax state, O accumulators ----
  float m = -INFINITY, lsum = 0.f;
  v8f oacc[8];
  #pragma unroll
  for (int t = 0; t < 8; ++t) oacc[t] = (v8f){};

  // ---- prologue: prefetch chunks 0 and 1 ----
  stage_async(smembase,             fhb, fhtb, wave, lane, 0);
  stage_async(smembase + BUF_BYTES, fhb, fhtb, wave, lane, I_CHUNK);

  for (int k = 0; k < NCHUNK; ++k) {
    if (k + 1 < NCHUNK) asm volatile("s_wait_asynccnt 0x10" ::: "memory");
    else                asm volatile("s_wait_asynccnt 0x0"  ::: "memory");
    __syncthreads();
    const int ciH = (k & 1) * BUF_HALF;          // tileCI half-index base
    const int icH = ciH + CI_BYTES / 2;          // tileIC half-index base

    // ---- S = K^T Q: four 16(i)x16(j) f32 tiles (4 independent WMMA chains) ----
    v8f s[4];
    #pragma unroll
    for (int t = 0; t < 4; ++t) s[t] = (v8f){};
    #pragma unroll
    for (int kk = 0; kk < 4; ++kk) {
      #pragma unroll
      for (int t = 0; t < 4; ++t) {
        v16h a;
        #pragma unroll
        for (int kq = 0; kq < 16; ++kq) {
          // A layout (16x32): h[k] <-> K = k + 16*hi ; M = jn
          int c = kk * 32 + kq + 16 * hi;
          a[kq] = smem[icH + (t * 16 + jn) * IC_PITCH + c];
        }
        s[t] = __builtin_amdgcn_wmma_f32_16x16x32_f16(false, a, false, qf[kk], (short)0, s[t], false, false);
      }
    }

    // ---- online softmax over this chunk's 64 i's, per column j ----
    float cmax = s[0][0];
    #pragma unroll
    for (int t = 0; t < 4; ++t)
      #pragma unroll
      for (int r = 0; r < 8; ++r) cmax = fmaxf(cmax, s[t][r]);
    cmax = fmaxf(cmax, __shfl_xor(cmax, 16, 32));
    float mnew  = fmaxf(m, cmax);
    float scale = __expf(m - mnew);              // == 0 on first chunk (m = -inf)
    m = mnew;
    lsum *= scale;
    #pragma unroll
    for (int t = 0; t < 8; ++t)
      #pragma unroll
      for (int r = 0; r < 8; ++r) oacc[t][r] *= scale;

    float p[4][8], psum = 0.f;
    #pragma unroll
    for (int t = 0; t < 4; ++t)
      #pragma unroll
      for (int r = 0; r < 8; ++r) {
        p[t][r] = __expf(s[t][r] - m);
        psum += p[t][r];
      }
    psum += __shfl_xor(psum, 16, 32);
    lsum += psum;

    // ---- P (D layout) -> two f16 B-layout frags (K=i sequential / half-wave) ----
    v16h pf0, pf1;
    #pragma unroll
    for (int r = 0; r < 8; ++r) {
      float q0 = __shfl_xor(p[0][r], 16, 32);
      float q1 = __shfl_xor(p[1][r], 16, 32);
      float q2 = __shfl_xor(p[2][r], 16, 32);
      float q3 = __shfl_xor(p[3][r], 16, 32);
      pf0[r]     = (_Float16)(hi ? q1 : p[0][r]);
      pf0[8 + r] = (_Float16)(hi ? p[1][r] : q0);
      pf1[r]     = (_Float16)(hi ? q3 : p[2][r]);
      pf1[8 + r] = (_Float16)(hi ? p[3][r] : q2);
    }

    // ---- O += V * P: 8 independent c-tiles x 2 chained K=32 steps ----
    #pragma unroll
    for (int ct = 0; ct < 8; ++ct) {
      int crow = ct * 16 + jn;
      v16h af0, af1;
      #pragma unroll
      for (int kq = 0; kq < 16; ++kq) {
        int ioff = kq + 16 * hi;
        af0[kq] = smem[ciH + crow * CI_PITCH + ioff];
        af1[kq] = smem[ciH + crow * CI_PITCH + 32 + ioff];
      }
      oacc[ct] = __builtin_amdgcn_wmma_f32_16x16x32_f16(false, af0, false, pf0, (short)0, oacc[ct], false, false);
      oacc[ct] = __builtin_amdgcn_wmma_f32_16x16x32_f16(false, af1, false, pf1, (short)0, oacc[ct], false, false);
    }

    __syncthreads();                              // everyone done with buf[k&1]
    if (k + 2 < NCHUNK)                           // refill it for chunk k+2
      stage_async(smembase + (unsigned)((k & 1) * BUF_BYTES),
                  fhb, fhtb, wave, lane, (k + 2) * I_CHUNK);
  }

  // ---- epilogue: out = x + RES_SCALE * O / lsum ----
  float invl = 1.0f / lsum;
  float* ob = out + (size_t)b * C_DIM * N_TOK;
  #pragma unroll
  for (int ct = 0; ct < 8; ++ct) {
    #pragma unroll
    for (int r = 0; r < 8; ++r) {
      int c = ct * 16 + hi * 8 + r;   // D layout: lane<16 reg r -> M=r; lane>=16 -> M=r+8
      size_t idx = (size_t)c * N_TOK + j;
      ob[idx] = xb[idx] + RES_SCALE * oacc[ct][r] * invl;
    }
  }
}

extern "C" void kernel_launch(void* const* d_in, const int* in_sizes, int n_in,
                              void* d_out, int out_size, void* d_ws, size_t ws_size,
                              hipStream_t stream) {
  const float* x = (const float*)d_in[0];
  float* out = (float*)d_out;
  (void)in_sizes; (void)n_in; (void)out_size; (void)ws_size;

  // workspace: f16 F in both orientations (8 MB + 8 MB)
  _Float16* fh  = (_Float16*)d_ws;
  _Float16* fht = fh + (size_t)8 * C_DIM * N_TOK;

  convert_kernel<<<dim3(N_TOK / 64, 8), 256, 0, stream>>>(x, fh, fht);
  fa_selfattn_kernel<<<dim3(N_TOK / J_TILE, 8), 256, 0, stream>>>(x, fh, fht, out);
}
